// GCN_34823594836259
// MI455X (gfx1250) — compile-verified, matrix operand order checked by reading, not run on hardware
//
#include <hip/hip_runtime.h>
#include <hip/hip_bf16.h>

typedef __attribute__((ext_vector_type(16))) _Float16 v16h;
typedef __attribute__((ext_vector_type(8)))  _Float16 v8h;
typedef __attribute__((ext_vector_type(8)))  float    v8f;

// ---------------------------------------------------------------------------
// Small utility kernels
// ---------------------------------------------------------------------------
__global__ __launch_bounds__(256) void fill_val(float* p, int n, float v) {
    int i = blockIdx.x * 256 + threadIdx.x;
    if (i < n) p[i] = v;
}

__global__ __launch_bounds__(256) void zero_buf(float* p, size_t n) {
    size_t i = (size_t)blockIdx.x * 256 + threadIdx.x;
    if (i < n) p[i] = 0.0f;
}

__global__ __launch_bounds__(256) void count_deg(const int* __restrict__ dst,
                                                 float* __restrict__ deg, int E) {
    int i = blockIdx.x * 256 + threadIdx.x;
    if (i < E) atomicAdd(&deg[dst[i]], 1.0f);
}

__global__ __launch_bounds__(256) void rsqrt_ip(float* p, int n) {
    int i = blockIdx.x * 256 + threadIdx.x;
    if (i < n) p[i] = rsqrtf(p[i]);   // deg >= 1 always (self loop)
}

// ---------------------------------------------------------------------------
// WMMA GEMM:  C[M x NCOLS] = A[M x 128] @ W[128 x NCOLS]   (NCOLS <= NP)
// A fp32 -> f16 into LDS; W staged transposed (f16) so B-fragments are
// contiguous 16B LDS loads. f32 accumulation via v_wmma_f32_16x16x32_f16.
// Block = 256 threads = 8 waves; block tile 128 x NP. Each wave owns a
// 16-row strip; A-fragments (kt=0..3) hoisted out of the column-tile loop.
// NCOLS is compile-time so stores become base + immediate-offset.
// ---------------------------------------------------------------------------
template <int NP, int NCOLS>
__global__ __launch_bounds__(256) void gemm_wmma(const float* __restrict__ A,
                                                 const float* __restrict__ W,
                                                 float* __restrict__ C,
                                                 int M) {
    __shared__ alignas(16) _Float16 As[128 * 128];   // 32 KB
    __shared__ alignas(16) _Float16 Wt[NP * 128];    // NP=128 -> 32 KB

    const int tid  = threadIdx.x;
    const int row0 = blockIdx.x * 128;
    const bool full = (row0 + 128 <= M);             // 781 of 782 blocks

    // Stage A tile (128 x 128 fp32 -> f16), float4-vectorized.
    if (full) {
#pragma unroll
        for (int it = 0; it < 16; ++it) {
            int idx = tid + it * 256;                // 0..4095
            int r  = idx >> 5;
            int c4 = idx & 31;
            float4 v = ((const float4*)A)[(size_t)(row0 + r) * 32 + c4];
            _Float16* p = &As[r * 128 + c4 * 4];
            p[0] = (_Float16)v.x; p[1] = (_Float16)v.y;
            p[2] = (_Float16)v.z; p[3] = (_Float16)v.w;
        }
    } else {
        for (int idx = tid; idx < 128 * 32; idx += 256) {
            int r  = idx >> 5;
            int c4 = idx & 31;
            int gr = row0 + r;
            float4 v = make_float4(0.f, 0.f, 0.f, 0.f);
            if (gr < M) v = ((const float4*)A)[(size_t)gr * 32 + c4];
            _Float16* p = &As[r * 128 + c4 * 4];
            p[0] = (_Float16)v.x; p[1] = (_Float16)v.y;
            p[2] = (_Float16)v.z; p[3] = (_Float16)v.w;
        }
    }
    // Stage W transposed: Wt[n*128 + k] = W[k*NCOLS + n], zero-pad n >= NCOLS.
    for (int idx = tid; idx < NP * 128; idx += 256) {
        int n = idx >> 7;
        int k = idx & 127;
        float v = (n < NCOLS) ? W[k * NCOLS + n] : 0.0f;
        Wt[n * 128 + k] = (_Float16)v;
    }
    __syncthreads();

    const int wave = tid >> 5;
    const int lane = tid & 31;
    const int l15  = lane & 15;
    const int hi8  = (lane >= 16) ? 8 : 0;   // K offset for A/B frags, M offset for D
    const int m    = wave * 16 + l15;        // A-fragment row inside block tile

    // Hoist the 4 A-fragments (one per K step) out of the column-tile loop.
    v16h afrag[4];
#pragma unroll
    for (int kt = 0; kt < 4; ++kt) {
        const int kb = kt * 32 + hi8;
        // ISA 16-bit A layout: halves 0..7 = K kb..kb+7, halves 8..15 = K kb+16..kb+23
        v8h alo = *(const v8h*)&As[m * 128 + kb];
        v8h ahi = *(const v8h*)&As[m * 128 + kb + 16];
        afrag[kt] = __builtin_shufflevector(alo, ahi,
                        0,1,2,3,4,5,6,7,8,9,10,11,12,13,14,15);
    }

#pragma unroll
    for (int nt = 0; nt < NP / 16; ++nt) {
        v8f acc = {};
        const int n = nt * 16 + l15;         // B-fragment column
#pragma unroll
        for (int kt = 0; kt < 4; ++kt) {     // K = 128 in steps of 32
            const int kb = kt * 32 + hi8;
            v8h blo = *(const v8h*)&Wt[n * 128 + kb];
            v8h bhi = *(const v8h*)&Wt[n * 128 + kb + 16];
            v16h b = __builtin_shufflevector(blo, bhi,
                         0,1,2,3,4,5,6,7,8,9,10,11,12,13,14,15);
            acc = __builtin_amdgcn_wmma_f32_16x16x32_f16(
                false, afrag[kt], false, b, (short)0, acc, false, false);
        }
        // D layout: VGPR r -> tile row r + hi8, col = l15.
        const int col = nt * 16 + l15;
        const bool colOk = (NP == NCOLS) || (col < NCOLS);
        float* cp = C + (size_t)(row0 + wave * 16 + hi8) * NCOLS + col;
        if (full) {
            if (colOk) {
#pragma unroll
                for (int r = 0; r < 8; ++r)
                    cp[(size_t)r * NCOLS] = acc[r];   // immediate offsets
            }
        } else {
#pragma unroll
            for (int r = 0; r < 8; ++r) {
                int grow = row0 + wave * 16 + r + hi8;
                if (grow < M && colOk)
                    cp[(size_t)r * NCOLS] = acc[r];
            }
        }
    }
}

// ---------------------------------------------------------------------------
// Edge scatter: one wave per edge. agg[dst] += dinv[src]*dinv[dst] * H[src]
// ---------------------------------------------------------------------------
template <int F>
__global__ __launch_bounds__(256) void scatter_edges(const float* __restrict__ H,
                                                     const int* __restrict__ src,
                                                     const int* __restrict__ dst,
                                                     const float* __restrict__ dinv,
                                                     float* __restrict__ agg, int E) {
    int e = blockIdx.x * 8 + (threadIdx.x >> 5);
    if (e >= E) return;
    int lane = threadIdx.x & 31;
    int s = src[e], d = dst[e];
    float w = dinv[s] * dinv[d];
    if (lane * 4 < F) {
        const float4 v = ((const float4*)(H + (size_t)s * F))[lane];
        float* ad = agg + (size_t)d * F + lane * 4;
        atomicAdd(ad + 0, w * v.x);
        atomicAdd(ad + 1, w * v.y);
        atomicAdd(ad + 2, w * v.z);
        atomicAdd(ad + 3, w * v.w);
    }
}

// ---------------------------------------------------------------------------
// Finalize: out = act(agg + dinv^2 * H (self-loop) + bias)
// ACT: 0 = ReLU, 1 = sigmoid
// ---------------------------------------------------------------------------
template <int F, int ACT>
__global__ __launch_bounds__(256) void finalize_k(const float* __restrict__ agg,
                                                  const float* __restrict__ H,
                                                  const float* __restrict__ dinv,
                                                  const float* __restrict__ b,
                                                  float* __restrict__ out, int Nn) {
    size_t i = (size_t)blockIdx.x * 256 + threadIdx.x;
    if (i >= (size_t)Nn * F) return;
    int n = (int)(i / F);
    int f = (int)(i % F);
    float di = dinv[n];
    float v  = agg[i] + di * di * H[i] + b[f];
    out[i] = (ACT == 0) ? fmaxf(v, 0.0f) : (1.0f / (1.0f + expf(-v)));
}

// ---------------------------------------------------------------------------
// Driver
// ---------------------------------------------------------------------------
extern "C" void kernel_launch(void* const* d_in, const int* in_sizes, int n_in,
                              void* d_out, int out_size, void* d_ws, size_t ws_size,
                              hipStream_t stream) {
    const float* x  = (const float*)d_in[0];
    const int*   ei = (const int*)  d_in[1];
    const float* W0 = (const float*)d_in[2];
    const float* b0 = (const float*)d_in[3];
    const float* W1 = (const float*)d_in[4];
    const float* b1 = (const float*)d_in[5];
    const float* W2 = (const float*)d_in[6];
    const float* b2 = (const float*)d_in[7];
    float* out = (float*)d_out;

    const int Nn = in_sizes[0] / 128;    // 100000
    const int E  = in_sizes[1] / 2;      // 1600000
    const int* src = ei;                 // edge_index[0]
    const int* dst = ei + E;             // edge_index[1]

    // Workspace layout (floats): dinv | H | agg | hact  (~154 MB total)
    float* ws   = (float*)d_ws;
    float* dinv = ws;
    float* H    = ws + (((size_t)Nn + 63) & ~(size_t)63);
    float* agg  = H + (size_t)Nn * 128;
    float* hact = agg + (size_t)Nn * 128;

    const int gN  = (Nn + 255) / 256;
    const int gE  = (E + 255) / 256;
    const int gNF = (int)(((size_t)Nn * 128 + 255) / 256);
    const int gNC = (int)(((size_t)Nn * 40 + 255) / 256);
    const int gM  = (Nn + 127) / 128;
    const int gEw = (E + 7) / 8;

    // Degrees (self-loop => start at 1.0) and dinv = rsqrt(deg)
    fill_val<<<gN, 256, 0, stream>>>(dinv, Nn, 1.0f);
    count_deg<<<gE, 256, 0, stream>>>(dst, dinv, E);
    rsqrt_ip<<<gN, 256, 0, stream>>>(dinv, Nn);

    // Layer 0: H = x @ W0 ; aggregate ; relu -> hact
    gemm_wmma<128, 128><<<gM, 256, 0, stream>>>(x, W0, H, Nn);
    zero_buf<<<gNF, 256, 0, stream>>>(agg, (size_t)Nn * 128);
    scatter_edges<128><<<gEw, 256, 0, stream>>>(H, src, dst, dinv, agg, E);
    finalize_k<128, 0><<<gNF, 256, 0, stream>>>(agg, H, dinv, b0, hact, Nn);

    // Layer 1
    gemm_wmma<128, 128><<<gM, 256, 0, stream>>>(hact, W1, H, Nn);
    zero_buf<<<gNF, 256, 0, stream>>>(agg, (size_t)Nn * 128);
    scatter_edges<128><<<gEw, 256, 0, stream>>>(H, src, dst, dinv, agg, E);
    finalize_k<128, 0><<<gNF, 256, 0, stream>>>(agg, H, dinv, b1, hact, Nn);

    // Layer 2: 128 -> 40 (padded to 48 tile cols), sigmoid -> d_out
    gemm_wmma<48, 40><<<gM, 256, 0, stream>>>(hact, W2, H, Nn);
    zero_buf<<<gNC, 256, 0, stream>>>(agg, (size_t)Nn * 40);
    scatter_edges<40><<<gEw, 256, 0, stream>>>(H, src, dst, dinv, agg, E);
    finalize_k<40, 1><<<gNC, 256, 0, stream>>>(agg, H, dinv, b2, out, Nn);
}
